// Eucl_Mat_15015205667290
// MI455X (gfx1250) — matrix operand order repaired and checked by reading, not compiled, not run on hardware
//
#include <hip/hip_runtime.h>
#include <math.h>

// Pairwise Euclidean distance, N=8192, D=512, fp32 in/out.
//
// Pass 1: pre-split A into bf16 (hi + residual-lo) planes in d_ws, laid out
//         per k-chunk as [8192 rows][128B: 64B hi | 64B lo]  (16 MB total),
//         plus row squared-norms sq[8192] (32 KB).
// Pass 2: symmetric 128x128-tiled A*A^T via V_WMMA_F32_16X16X32_BF16 with
//         3 products per tile (hi*hi + hi*lo + lo*hi) ~ fp32 accuracy at
//         bf16-WMMA throughput. Tiles are staged into double-buffered LDS by
//         the Tensor Data Mover (tensor_load_to_lds) with D# pad_enable
//         reproducing the 144B bank-conflict-free row stride. Upper-triangle
//         blocks only (2080); the mirror tile is stored via an LDS transpose.

#define NROWS 8192
#define DDIM  512
#define BK    32
#define NKT   (DDIM / BK)                // 16 k-steps
#define ROWB  144                        // LDS row: 64B hi + 64B lo + 16B pad
#define TILE_STRIDE (128 * ROWB)         // 18432 B per staged 128-row tile
#define BUF_STRIDE  (2 * TILE_STRIDE)    // A-tile + B-tile
#define STAGE_BYTES (2 * BUF_STRIDE)     // double buffered = 73728 B
#define PADW  132                        // result tile pitch (floats)
#define SQ_OFF STAGE_BYTES
#define LDS_BYTES (STAGE_BYTES + 256 * 4)   // 74752 B; CDNA5 allows 320KB/WG

#define WS_SQ_BYTES (NROWS * 4)                       // sq[] first in d_ws
#define WS_PLANE    ((size_t)NROWS * 128)             // 1 MB per k-chunk
#define WS_BF_OFF   ((size_t)WS_SQ_BYTES)             // planes after sq

typedef __bf16 bf16_t;
typedef bf16_t v16bf __attribute__((ext_vector_type(16)));
typedef bf16_t v8bf  __attribute__((ext_vector_type(8)));
typedef float  v8f   __attribute__((ext_vector_type(8)));
typedef unsigned int v4u __attribute__((ext_vector_type(4)));
typedef unsigned int v8u __attribute__((ext_vector_type(8)));

// Two ds_load_b128 from LDS -> one 16xbf16 WMMA fragment.
static __device__ __forceinline__ v16bf frag_ld(const unsigned char* p, int o0, int o1) {
  v8bf a = *(const v8bf*)(p + o0);
  v8bf b = *(const v8bf*)(p + o1);
  return __builtin_shufflevector(a, b, 0,1,2,3,4,5,6,7,8,9,10,11,12,13,14,15);
}

// Split two fp32 into packed bf16-hi pair and bf16-lo (residual) pair.
static __device__ __forceinline__ void split2(float x0, float x1, unsigned &hp, unsigned &lp) {
  unsigned u0 = __float_as_uint(x0), u1 = __float_as_uint(x1);
  unsigned h0 = u0 & 0xffff0000u,  h1 = u1 & 0xffff0000u;
  float r0 = x0 - __uint_as_float(h0);
  float r1 = x1 - __uint_as_float(h1);
  hp = (h0 >> 16) | h1;
  lp = (__float_as_uint(r0) >> 16) | (__float_as_uint(r1) & 0xffff0000u);
}

// TDM: one 2-D tile (32 dwords x 128 rows) from global -> LDS, inserting
// 4 dwords of LDS padding after every 32 dwords (row stride 128B -> 144B).
// Groups 2/3 omitted (NULL) => up-to-2D tensor. Issued once per wave.
static __device__ __forceinline__ void tdm_load_tile(unsigned lds_addr,
                                                     unsigned long long gaddr) {
  v4u g0;
  g0.x = 1u;                                   // count=1 (valid), load, no gather
  g0.y = lds_addr;                             // LDS byte address
  g0.z = (unsigned)gaddr;                      // global_addr[31:0]
  g0.w = (unsigned)((gaddr >> 32) & 0x1FFFFFFu) | (2u << 30);  // addr[56:32] | type=2
  v8u g1;
  g1.s0 = (2u << 16)      // data_size = 4 bytes
        | (1u << 20)      // pad_enable
        | (4u << 22)      // pad_interval: 2^4*2 = 32 dwords
        | (3u << 25);     // pad_amount: 4 dwords (16B)
  g1.s1 = (32u  << 16);   // tensor_dim0 = 32 dwords (low16 -> bits[63:48])
  g1.s2 = (128u << 16);   // tensor_dim0 hi=0 | tensor_dim1 = 128 rows
  g1.s3 = (32u  << 16);   // tensor_dim1 hi=0 | tile_dim0 = 32 dwords
  g1.s4 = 128u;           // tile_dim1 = 128 rows, tile_dim2 = 0
  g1.s5 = 32u;            // tensor_dim0_stride = 32 dwords (contiguous rows)
  g1.s6 = 0u;             // stride0 hi | tensor_dim1_stride lo (unused for 2D)
  g1.s7 = 0u;
  asm volatile("tensor_load_to_lds %0, %1" :: "s"(g0), "s"(g1) : "memory");
}

// -------- Kernel 1a: row squared-norms -> d_ws[0 .. 8192) floats ----------
__launch_bounds__(256)
__global__ void rownorm_kernel(const float* __restrict__ A, float* __restrict__ sq) {
  int row  = blockIdx.x * 8 + (threadIdx.x >> 5);   // one wave32 per row
  int lane = threadIdx.x & 31;
  const float4* p = (const float4*)(A + (size_t)row * DDIM);
  float s = 0.f;
#pragma unroll
  for (int j = 0; j < 4; ++j) {
    float4 v = p[lane + 32 * j];
    s += v.x * v.x + v.y * v.y + v.z * v.z + v.w * v.w;
  }
#pragma unroll
  for (int o = 16; o; o >>= 1) s += __shfl_xor(s, o, 32);
  if (lane == 0) sq[row] = s;
}

// -------- Kernel 1b: pre-split A into bf16 hi/lo planes in d_ws ----------
// Plane layout (per kc): row r at byte r*128: [64B hi (32 bf16) | 64B lo].
__launch_bounds__(256)
__global__ void presplit_kernel(const float* __restrict__ A,
                                unsigned char* __restrict__ wsb) {
  int e  = blockIdx.x * 256 + threadIdx.x;   // 2^20 threads, 4 floats each
  int q  = e & 7;                            // float4 index within 32-float chunk
  int r  = (e >> 3) & (NROWS - 1);           // row
  int kc = e >> 16;                          // k-chunk
  float4 v = *(const float4*)(A + (size_t)r * DDIM + kc * BK + 4 * q);
  unsigned h0, l0, h1, l1;
  split2(v.x, v.y, h0, l0);
  split2(v.z, v.w, h1, l1);
  unsigned char* rowp = wsb + (size_t)kc * WS_PLANE + (size_t)r * 128;
  *(unsigned long long*)(rowp + 8 * q)      = h0 | ((unsigned long long)h1 << 32);
  *(unsigned long long*)(rowp + 64 + 8 * q) = l0 | ((unsigned long long)l1 << 32);
}

// -------- Kernel 2: TDM-staged symmetric distance GEMM ----------
__launch_bounds__(256)
__global__ void dist_kernel(const unsigned char* __restrict__ wsb,
                            const float* __restrict__ sq,
                            float* __restrict__ out) {
  extern __shared__ unsigned char smem[];
  const int tid  = threadIdx.x;
  const int lane = tid & 31;
  const int wave = tid >> 5;        // 8 waves
  const int wm   = wave & 3;        // wave row-block (4 x 32 rows)
  const int wn   = wave >> 2;       // wave col-block (2 x 64 cols)
  const int l16  = lane & 15;
  const int hi16 = lane >> 4;

  // Decode upper-triangle tile pair: t -> (c, r) with r >= c.
  unsigned t = blockIdx.x;
  unsigned r = (unsigned)((sqrtf(8.0f * (float)t + 1.0f) - 1.0f) * 0.5f);
  while ((r + 1) * (r + 2) / 2 <= t) ++r;
  while (r * (r + 1) / 2 > t) --r;
  unsigned c = t - r * (r + 1) / 2;
  const int i0 = (int)c * 128;   // output row block
  const int j0 = (int)r * 128;   // output col block (>= i0)

  float* sqRow = (float*)(smem + SQ_OFF);
  float* sqCol = sqRow + 128;
  if (tid < 128) sqRow[tid] = sq[i0 + tid];
  else           sqCol[tid - 128] = sq[j0 + tid - 128];

  const unsigned lds_base = (unsigned)(uintptr_t)(void*)smem;  // LDS byte offset
  const unsigned long long gA = (unsigned long long)(uintptr_t)wsb + (size_t)i0 * 128;
  const unsigned long long gB = (unsigned long long)(uintptr_t)wsb + (size_t)j0 * 128;

  v8f acc[2][4] = {};

  // Issue both tile DMAs for k-chunk kc into buffer buf (wave 0 only).
  auto stage = [&](int kc, int buf) {
    unsigned ldst = lds_base + buf * BUF_STRIDE;
    unsigned long long goff = (unsigned long long)kc * WS_PLANE;
    tdm_load_tile(ldst,               gA + goff);
    tdm_load_tile(ldst + TILE_STRIDE, gB + goff);
  };

  // One K-chunk: 12 fragment fetches + 24 v_wmma_f32_16x16x32_bf16 per wave.
  auto computeK = [&](int kc) {
    const unsigned char* base = smem + (kc & 1) * BUF_STRIDE;
    const int ga = hi16 * 16;   // A frag: lanes<16: K0-7 & K16-23; lanes>=16: K8-15 & K24-31
    const int gb = hi16 * 16;   // B frag: lanes<16: K0-7 & K16-23; lanes>=16: K8-15 & K24-31
    v16bf ah[2], al[2];
#pragma unroll
    for (int m = 0; m < 2; ++m) {
      const unsigned char* p = base + (32 * wm + 16 * m + l16) * ROWB;
      ah[m] = frag_ld(p,      ga, ga + 32);
      al[m] = frag_ld(p + 64, ga, ga + 32);
    }
#pragma unroll
    for (int n = 0; n < 4; ++n) {
      const unsigned char* p = base + TILE_STRIDE + (64 * wn + 16 * n + l16) * ROWB;
      v16bf bh = frag_ld(p,      gb, gb + 32);
      v16bf bl = frag_ld(p + 64, gb, gb + 32);
#pragma unroll
      for (int m = 0; m < 2; ++m) {
        acc[m][n] = __builtin_amdgcn_wmma_f32_16x16x32_bf16(false, ah[m], false, bh,
                                                            (short)0, acc[m][n], false, false);
        acc[m][n] = __builtin_amdgcn_wmma_f32_16x16x32_bf16(false, ah[m], false, bl,
                                                            (short)0, acc[m][n], false, false);
        acc[m][n] = __builtin_amdgcn_wmma_f32_16x16x32_bf16(false, al[m], false, bh,
                                                            (short)0, acc[m][n], false, false);
      }
    }
  };

  // Pipeline: TDM stages k+1 while all waves run WMMA on k.
  if (wave == 0) {
    stage(0, 0);
    __builtin_amdgcn_s_wait_tensorcnt(0);
  }
  __syncthreads();
  for (int kc = 0; kc < NKT; ++kc) {
    if (wave == 0 && kc + 1 < NKT) stage(kc + 1, (kc + 1) & 1);
    computeK(kc);
    if (kc + 1 < NKT) {
      if (wave == 0) __builtin_amdgcn_s_wait_tensorcnt(0);
      __syncthreads();
    }
  }
  __syncthreads();   // staging LDS is reused as the result tile below

  // Epilogue: d = sqrt(max(sq_i + sq_j - 2*gram, 0)) -> padded LDS tile.
  float* res = (float*)smem;
#pragma unroll
  for (int m = 0; m < 2; ++m)
#pragma unroll
    for (int n = 0; n < 4; ++n) {
      int col = 64 * wn + 16 * n + l16;
      float sc = sqCol[col];
#pragma unroll
      for (int i = 0; i < 8; ++i) {
        int rl = 32 * wm + 16 * m + i + hi16 * 8;   // C layout: VGPR i -> M=i (+8 high lanes)
        float d2 = sqRow[rl] + sc - 2.0f * acc[m][n][i];
        d2 = d2 > 0.f ? d2 : 0.f;
        res[rl * PADW + col] = __builtin_sqrtf(d2);
      }
    }
  __syncthreads();

  // Store the tile (coalesced float4).
#pragma unroll 4
  for (int e = tid; e < 4096; e += 256) {
    int m = e >> 5, n4 = e & 31;
    float4 v = *(const float4*)(res + m * PADW + 4 * n4);
    *(float4*)(out + (size_t)(i0 + m) * NROWS + j0 + 4 * n4) = v;
  }
  // Mirrored tile via LDS transpose reads (still coalesced float4 stores).
  if (i0 != j0) {
#pragma unroll 4
    for (int e = tid; e < 4096; e += 256) {
      int nn = e >> 5, mm4 = e & 31;
      float4 v;
      v.x = res[(4 * mm4 + 0) * PADW + nn];
      v.y = res[(4 * mm4 + 1) * PADW + nn];
      v.z = res[(4 * mm4 + 2) * PADW + nn];
      v.w = res[(4 * mm4 + 3) * PADW + nn];
      *(float4*)(out + (size_t)(j0 + nn) * NROWS + i0 + 4 * mm4) = v;
    }
  }
}

extern "C" void kernel_launch(void* const* d_in, const int* in_sizes, int n_in,
                              void* d_out, int out_size, void* d_ws, size_t ws_size,
                              hipStream_t stream) {
  const float* A = (const float*)d_in[0];
  float* out = (float*)d_out;
  // d_ws layout: [0, 32KB) sq[8192]; [32KB, 32KB+16MB) bf16 hi/lo planes.
  float* sq = (float*)d_ws;
  unsigned char* wsb = (unsigned char*)d_ws + WS_BF_OFF;

  rownorm_kernel<<<NROWS / 8, 256, 0, stream>>>(A, sq);
  presplit_kernel<<<(NROWS * DDIM / 4) / 256, 256, 0, stream>>>(A, wsb);
  dist_kernel<<<(64 * 65) / 2, 256, LDS_BYTES, stream>>>(wsb, sq, out);
  (void)in_sizes; (void)n_in; (void)out_size; (void)ws_size;
}
